// FlashAttentionFusedAttention_28656021799473
// MI455X (gfx1250) — matrix operand order repaired and checked by reading, not compiled, not run on hardware
//
#include <hip/hip_runtime.h>
#include <cmath>

// ---------------------------------------------------------------------------
// CDNA5 (gfx1250) fused attention block:
//   Q/KV projections (bf16 WMMA) -> RoPE -> GQA causal flash attention
//   (bf16 WMMA, online softmax) -> output projection (bf16 WMMA, f32 out).
// All matrix math uses v_wmma_f32_16x16x32_bf16 with f32 accumulation.
// GEMM: 32x64 tile per wave -> 8 WMMA per 12 b128 loads per k-step.
// ---------------------------------------------------------------------------

typedef __attribute__((ext_vector_type(16))) __bf16 v16bf;
typedef __attribute__((ext_vector_type(8)))  float  v8f;

union BFrag16 { v16bf v; uint4 u[2]; };

constexpr int Bsz  = 2;
constexpr int S    = 2048;
constexpr int H    = 2048;
constexpr int NH   = 16;
constexpr int NKV  = 4;
constexpr int D    = 128;
constexpr int NREP = NH / NKV;

// ---------------- elementwise converts ----------------

__global__ void f32_to_bf16_kernel(const float* __restrict__ in,
                                   __bf16* __restrict__ out, int n) {
  int i = blockIdx.x * blockDim.x + threadIdx.x;
  if (i < n) out[i] = (__bf16)in[i];
}

// in: (K,N) f32 row-major  ->  out: (N,K) bf16 row-major (transposed)
__global__ void transpose_f32_to_bf16_kernel(const float* __restrict__ in,
                                             __bf16* __restrict__ out,
                                             int K, int N) {
  int i = blockIdx.x * blockDim.x + threadIdx.x;
  if (i < K * N) {
    int k = i / N, n = i % N;
    out[(size_t)n * K + k] = (__bf16)in[i];
  }
}

// ---------------- bf16 GEMM: C[M,N](f32) = A[M,K] * Bt[N,K]^T ----------------
// One wave computes a 32x64 tile (2 row-fragments x 4 col-tiles -> 8 WMMA per
// 32-wide k-step, each B-fragment reused by both row-fragments).
// blockDim=256 (8 waves); wave w covers columns blockIdx.y*512 + w*64.
// M%32==0, N%512==0, K%32==0 assumed.

__global__ __launch_bounds__(256)
void gemm_bf16_kernel(const __bf16* __restrict__ A, const __bf16* __restrict__ Bt,
                      float* __restrict__ C, int M, int N, int K) {
  const int lane = threadIdx.x & 31;
  const int wave = threadIdx.x >> 5;
  const int half = lane >> 4;   // 0: lanes 0-15, 1: lanes 16-31
  const int ln   = lane & 15;
  const int m0 = blockIdx.x * 32;
  const int n0 = blockIdx.y * 512 + wave * 64;

  v8f acc[2][4];
  for (int mi = 0; mi < 2; ++mi)
    for (int t = 0; t < 4; ++t)
      for (int r = 0; r < 8; ++r) acc[mi][t][r] = 0.f;

  const __bf16* arow0 = A + (size_t)(m0 + ln) * K;
  const __bf16* arow1 = A + (size_t)(m0 + 16 + ln) * K;
  for (int k = 0; k < K; k += 32) {
    // A fragments (16x32 bf16): elems 0..7 -> K = k + 8*half + e,
    //                           elems 8..15 -> K = k + 16 + 8*half + (e-8)
    BFrag16 a0, a1;
    a0.u[0] = *(const uint4*)(arow0 + k + half * 8);
    a0.u[1] = *(const uint4*)(arow0 + k + 16 + half * 8);
    a1.u[0] = *(const uint4*)(arow1 + k + half * 8);
    a1.u[1] = *(const uint4*)(arow1 + k + 16 + half * 8);
    for (int t = 0; t < 4; ++t) {
      // B fragment (32x16): col = n, elems 0..7 -> K = k + 8*half + e,
      //                     elems 8..15 -> K = k + 16 + 8*half + (e-8)
      const __bf16* bcol = Bt + (size_t)(n0 + t * 16 + ln) * K + k + half * 8;
      BFrag16 b;
      b.u[0] = *(const uint4*)(bcol);
      b.u[1] = *(const uint4*)(bcol + 16);
      acc[0][t] = __builtin_amdgcn_wmma_f32_16x16x32_bf16(
          false, a0.v, false, b.v, (short)0, acc[0][t], false, false);
      acc[1][t] = __builtin_amdgcn_wmma_f32_16x16x32_bf16(
          false, a1.v, false, b.v, (short)0, acc[1][t], false, false);
    }
  }
  // C layout: VGPR r -> row m0 + mi*16 + r + 8*half, lane%16 -> col
  for (int mi = 0; mi < 2; ++mi)
    for (int t = 0; t < 4; ++t)
      for (int r = 0; r < 8; ++r)
        C[(size_t)(m0 + mi * 16 + r + half * 8) * N + n0 + t * 16 + ln] =
            acc[mi][t][r];
}

// ---------------- RoPE + layout shuffles ----------------

// Qf: (B,S,NH*D) f32 -> Qb: (B,NH,S,D) bf16 with rotate-half RoPE
__global__ void rope_q_kernel(const float* __restrict__ Qf,
                              const float* __restrict__ fcos,
                              const float* __restrict__ fsin,
                              __bf16* __restrict__ Qb) {
  const int hd = D / 2;
  int i = blockIdx.x * blockDim.x + threadIdx.x;
  if (i >= Bsz * S * NH * hd) return;
  int p = i % hd; int t = i / hd;
  int h = t % NH; t /= NH;
  int s = t % S;  int b = t / S;
  size_t src = ((size_t)(b * S + s) * NH + h) * D;
  float x1 = Qf[src + p], x2 = Qf[src + p + hd];
  float c1 = fcos[s * D + p],      s1 = fsin[s * D + p];
  float c2 = fcos[s * D + p + hd], s2 = fsin[s * D + p + hd];
  size_t dst = ((size_t)(b * NH + h) * S + s) * D;
  Qb[dst + p]      = (__bf16)(x1 * c1 - x2 * s1);
  Qb[dst + p + hd] = (__bf16)(x2 * c2 + x1 * s2);
}

// KVf: (B,S,2*NKV*D) f32 -> Kb: (B,NKV,S,D) bf16 (RoPE), Vt: (B,NKV,D,S) bf16
__global__ void rope_kv_kernel(const float* __restrict__ KVf,
                               const float* __restrict__ fcos,
                               const float* __restrict__ fsin,
                               __bf16* __restrict__ Kb,
                               __bf16* __restrict__ Vt) {
  const int hd = D / 2;
  int i = blockIdx.x * blockDim.x + threadIdx.x;
  if (i >= Bsz * S * NKV * hd) return;
  int p = i % hd; int t = i / hd;
  int h = t % NKV; t /= NKV;
  int s = t % S;   int b = t / S;
  size_t row = (size_t)(b * S + s) * (2 * NKV * D);
  float k1 = KVf[row + h * D + p], k2 = KVf[row + h * D + p + hd];
  float v1 = KVf[row + NKV * D + h * D + p];
  float v2 = KVf[row + NKV * D + h * D + p + hd];
  float c1 = fcos[s * D + p],      s1 = fsin[s * D + p];
  float c2 = fcos[s * D + p + hd], s2 = fsin[s * D + p + hd];
  size_t kdst = ((size_t)(b * NKV + h) * S + s) * D;
  Kb[kdst + p]      = (__bf16)(k1 * c1 - k2 * s1);
  Kb[kdst + p + hd] = (__bf16)(k2 * c2 + k1 * s2);
  size_t vdst = (size_t)(b * NKV + h) * D * S;
  Vt[vdst + (size_t)p * S + s]        = (__bf16)v1;
  Vt[vdst + (size_t)(p + hd) * S + s] = (__bf16)v2;
}

// ---------------- flash attention (1 wave / 16-query tile / head) ----------------
// scores = (Q K^T)/sqrt(D) with causal + key-padding mask, online softmax,
// O = P V. Qb:(B,NH,S,D), Kb:(B,NKV,S,D), Vt:(B,NKV,D,S), Ob:(B,S,NH*D) bf16.

__global__ __launch_bounds__(32)
void flash_attn_kernel(const __bf16* __restrict__ Qb, const __bf16* __restrict__ Kb,
                       const __bf16* __restrict__ Vt, const int* __restrict__ seqlen,
                       __bf16* __restrict__ Ob) {
  __shared__ __bf16 pshare[16 * 32];
  const int lane = threadIdx.x;
  const int half = lane >> 4;
  const int ln   = lane & 15;
  const int q0  = blockIdx.x * 16;
  const int h   = blockIdx.y;
  const int b   = blockIdx.z;
  const int kvh = h / NREP;
  const int sl  = seqlen[b];
  const float scale = 0.08838834764831845f;  // 1/sqrt(128)

  // Q tile (16x128) as 4 A-fragments, loaded once
  BFrag16 qa[4];
  const __bf16* qrow = Qb + ((size_t)(b * NH + h) * S + q0 + ln) * D;
  for (int kc = 0; kc < 4; ++kc) {
    qa[kc].u[0] = *(const uint4*)(qrow + kc * 32 + half * 8);
    qa[kc].u[1] = *(const uint4*)(qrow + kc * 32 + 16 + half * 8);
  }

  float mrow[8], lrow[8];
  v8f o[8];                       // O tile 16x128: col = dt*16+ln, row = r+8*half
  for (int r = 0; r < 8; ++r) { mrow[r] = -__builtin_inff(); lrow[r] = 0.f; }
  for (int dt = 0; dt < 8; ++dt)
    for (int r = 0; r < 8; ++r) o[dt][r] = 0.f;

  const __bf16* kb = Kb + (size_t)(b * NKV + kvh) * S * D;
  const __bf16* vb = Vt + (size_t)(b * NKV + kvh) * D * S;

  const int jend = q0 + 15;
  for (int j0 = 0; j0 <= jend; j0 += 32) {
    // ---- scores: two 16x16 tiles (keys j0..j0+15, j0+16..j0+31) ----
    v8f sc[2];
    for (int t = 0; t < 2; ++t) {
      for (int r = 0; r < 8; ++r) sc[t][r] = 0.f;
      const __bf16* krow = kb + (size_t)(j0 + t * 16 + ln) * D + half * 8;
      for (int kc = 0; kc < 4; ++kc) {
        BFrag16 kf;
        kf.u[0] = *(const uint4*)(krow + kc * 32);
        kf.u[1] = *(const uint4*)(krow + kc * 32 + 16);
        sc[t] = __builtin_amdgcn_wmma_f32_16x16x32_bf16(
            false, qa[kc].v, false, kf.v, (short)0, sc[t], false, false);
      }
    }
    // ---- scale + causal + seq-length mask ----
    for (int t = 0; t < 2; ++t) {
      int key = j0 + t * 16 + ln;
      bool kvalid = key < sl;
      for (int r = 0; r < 8; ++r) {
        int qr = q0 + r + half * 8;
        float v = sc[t][r] * scale;
        sc[t][r] = (kvalid && key <= qr) ? v : -__builtin_inff();
      }
    }
    // ---- online softmax (row reductions across 16-lane groups) ----
    float pnew[2][8];
    for (int r = 0; r < 8; ++r) {
      float bm = fmaxf(sc[0][r], sc[1][r]);
      for (int off = 8; off >= 1; off >>= 1)
        bm = fmaxf(bm, __shfl_xor(bm, off, 32));
      float mn   = fmaxf(mrow[r], bm);
      float corr = __expf(mrow[r] - mn);
      float p0   = __expf(sc[0][r] - mn);
      float p1   = __expf(sc[1][r] - mn);
      float ps = p0 + p1;
      for (int off = 8; off >= 1; off >>= 1)
        ps += __shfl_xor(ps, off, 32);
      lrow[r] = lrow[r] * corr + ps;
      mrow[r] = mn;
      pnew[0][r] = p0; pnew[1][r] = p1;
      for (int dt = 0; dt < 8; ++dt) o[dt][r] *= corr;
    }
    // ---- transpose P (C layout -> A layout) through LDS ----
    for (int t = 0; t < 2; ++t)
      for (int r = 0; r < 8; ++r)
        pshare[(r + half * 8) * 32 + t * 16 + ln] = (__bf16)pnew[t][r];
    __syncthreads();
    BFrag16 pa;
    pa.u[0] = *(const uint4*)(&pshare[ln * 32 + half * 8]);
    pa.u[1] = *(const uint4*)(&pshare[ln * 32 + 16 + half * 8]);
    // ---- O += P * V (contraction over 32 keys) ----
    for (int dt = 0; dt < 8; ++dt) {
      const __bf16* vrow = vb + (size_t)(dt * 16 + ln) * S + j0 + half * 8;
      BFrag16 vf;
      vf.u[0] = *(const uint4*)(vrow);
      vf.u[1] = *(const uint4*)(vrow + 16);
      o[dt] = __builtin_amdgcn_wmma_f32_16x16x32_bf16(
          false, pa.v, false, vf.v, (short)0, o[dt], false, false);
    }
    __syncthreads();
  }
  // ---- epilogue: O / l, write (B,S,NH*D) bf16 ----
  for (int dt = 0; dt < 8; ++dt)
    for (int r = 0; r < 8; ++r) {
      int row = q0 + r + half * 8;
      float val = o[dt][r] / lrow[r];
      Ob[((size_t)(b * S + row) * NH + h) * D + dt * 16 + ln] = (__bf16)val;
    }
}

// ---------------- host orchestration ----------------

extern "C" void kernel_launch(void* const* d_in, const int* in_sizes, int n_in,
                              void* d_out, int out_size, void* d_ws, size_t ws_size,
                              hipStream_t stream) {
  const float* x    = (const float*)d_in[0];
  const int*   slen = (const int*)d_in[1];
  const float* fcos = (const float*)d_in[2];
  const float* fsin = (const float*)d_in[3];
  const float* Wq   = (const float*)d_in[4];
  const float* Wkv  = (const float*)d_in[5];
  const float* Wo   = (const float*)d_in[6];
  float* out = (float*)d_out;

  char* ws = (char*)d_ws;
  size_t off = 0;
  auto alloc = [&](size_t bytes) -> void* {
    void* p = ws + off;
    off += (bytes + 255) & ~(size_t)255;
    return p;
  };

  const int M = Bsz * S;                       // 4096 rows for all GEMMs
  __bf16* xb    = (__bf16*)alloc((size_t)M * H * 2);
  __bf16* Wqt   = (__bf16*)alloc((size_t)(NH * D) * H * 2);
  __bf16* Wkvt  = (__bf16*)alloc((size_t)(2 * NKV * D) * H * 2);
  __bf16* Wot   = (__bf16*)alloc((size_t)H * H * 2);
  float*  Qf    = (float*) alloc((size_t)M * NH * D * 4);
  float*  KVf   = (float*) alloc((size_t)M * 2 * NKV * D * 4);
  __bf16* Qb    = (__bf16*)alloc((size_t)M * NH * D * 2);
  __bf16* Kb    = (__bf16*)alloc((size_t)Bsz * NKV * S * D * 2);
  __bf16* Vt    = (__bf16*)alloc((size_t)Bsz * NKV * D * S * 2);
  __bf16* attnb = (__bf16*)alloc((size_t)M * NH * D * 2);

  // 1) convert inputs to bf16 (weights transposed for B-fragment loads)
  {
    int n = M * H;
    f32_to_bf16_kernel<<<(n + 255) / 256, 256, 0, stream>>>(x, xb, n);
  }
  transpose_f32_to_bf16_kernel<<<(H * NH * D + 255) / 256, 256, 0, stream>>>(
      Wq, Wqt, H, NH * D);
  transpose_f32_to_bf16_kernel<<<(H * 2 * NKV * D + 255) / 256, 256, 0, stream>>>(
      Wkv, Wkvt, H, 2 * NKV * D);
  transpose_f32_to_bf16_kernel<<<(H * H + 255) / 256, 256, 0, stream>>>(
      Wo, Wot, H, H);

  // 2) projections
  gemm_bf16_kernel<<<dim3(M / 32, (NH * D) / 512), 256, 0, stream>>>(
      xb, Wqt, Qf, M, NH * D, H);
  gemm_bf16_kernel<<<dim3(M / 32, (2 * NKV * D) / 512), 256, 0, stream>>>(
      xb, Wkvt, KVf, M, 2 * NKV * D, H);

  // 3) RoPE + layout shuffles
  {
    int n = Bsz * S * NH * (D / 2);
    rope_q_kernel<<<(n + 255) / 256, 256, 0, stream>>>(Qf, fcos, fsin, Qb);
  }
  {
    int n = Bsz * S * NKV * (D / 2);
    rope_kv_kernel<<<(n + 255) / 256, 256, 0, stream>>>(KVf, fcos, fsin, Kb, Vt);
  }

  // 4) GQA causal flash attention
  flash_attn_kernel<<<dim3(S / 16, NH, Bsz), 32, 0, stream>>>(
      Qb, Kb, Vt, slen, attnb);

  // 5) output projection -> f32 d_out
  gemm_bf16_kernel<<<dim3(M / 32, H / 512), 256, 0, stream>>>(
      attnb, Wot, out, M, H, H);

  (void)in_sizes; (void)n_in; (void)out_size; (void)ws_size;
}
